// DRASIModel_47579647705649
// MI455X (gfx1250) — compile-verified, matrix operand order checked by reading, not run on hardware
//
#include <hip/hip_runtime.h>
#include <math.h>

typedef __attribute__((ext_vector_type(16))) __bf16 v16bf;
typedef __attribute__((ext_vector_type(8)))  __bf16 v8bf;
typedef __attribute__((ext_vector_type(8)))  float  v8f;

// ---------------------------------------------------------------------------
// fp32 <-> bf16: use native conversions (clang lowers to v_cvt_* on gfx1250)
// ---------------------------------------------------------------------------
__device__ __forceinline__ __bf16 f2bf(float f) { return static_cast<__bf16>(f); }
__device__ __forceinline__ float  bf2f(__bf16 b) { return static_cast<float>(b); }

// WMMA 16-bit A/B per-lane K pattern: VGPR v holds K pair starting at
//   kl(v, half) = (v<4 ? 2v : 16 + 2(v-4)) + 8*half
__device__ __forceinline__ int frag_k(int v, int pair, int half) {
    return ((v < 4) ? 2 * v : 16 + 2 * (v - 4)) + 8 * half + pair;
}

// ---------------------------------------------------------------------------
// A-fragment loaders (16x32 tile, per-lane = two contiguous 8-elem runs)
// ---------------------------------------------------------------------------
__device__ __forceinline__ v16bf load_a_frag(const __bf16* __restrict__ row,
                                             int koff, int half) {
    union { v16bf v; struct { v8bf lo, hi; } p; } u;
    u.p.lo = *(const v8bf*)(row + koff + half * 8);        // 16B load
    u.p.hi = *(const v8bf*)(row + koff + 16 + half * 8);   // 16B load
    return u.v;
}
__device__ __forceinline__ v16bf load_a_frag(const float* __restrict__ row,
                                             int koff, int half) {
    const float4 f0 = *(const float4*)(row + koff + half * 8);
    const float4 f1 = *(const float4*)(row + koff + half * 8 + 4);
    const float4 f2 = *(const float4*)(row + koff + 16 + half * 8);
    const float4 f3 = *(const float4*)(row + koff + 16 + half * 8 + 4);
    v16bf a;
    a[0]  = f2bf(f0.x); a[1]  = f2bf(f0.y); a[2]  = f2bf(f0.z); a[3]  = f2bf(f0.w);
    a[4]  = f2bf(f1.x); a[5]  = f2bf(f1.y); a[6]  = f2bf(f1.z); a[7]  = f2bf(f1.w);
    a[8]  = f2bf(f2.x); a[9]  = f2bf(f2.y); a[10] = f2bf(f2.z); a[11] = f2bf(f2.w);
    a[12] = f2bf(f3.x); a[13] = f2bf(f3.y); a[14] = f2bf(f3.z); a[15] = f2bf(f3.w);
    return a;
}

// ---------------------------------------------------------------------------
// Fused GEMM:  C[M,OUT] = act( A0[M,K0]@B0[K0,OUT] + A1[M,K1]@B1[K1,OUT] + bias )
// bf16 WMMA, fp32 accumulate. 256 threads = 8 waves, 128 rows/block,
// full OUT width per wave. Weights staged in LDS pre-swizzled into
// B-fragment layout (one contiguous 32B chunk per lane per fragment).
// Output either bf16 (activations) or f32 (externally visible tensors).
// ---------------------------------------------------------------------------
template<typename TA0, typename TA1, int K0, int K1, int OUT, bool RELU, bool OBF>
__global__ __launch_bounds__(256)
void gemm_kernel(const TA0* __restrict__ A0, const TA1* __restrict__ A1,
                 const float* __restrict__ B0, const float* __restrict__ B1,
                 const float* __restrict__ bias, void* __restrict__ Cout, int M)
{
    constexpr int K  = K0 + K1;
    constexpr int NT = OUT / 16;
    constexpr int KT = K / 32;
    constexpr int NFRAG = KT * NT;

    __shared__ v16bf sW[NFRAG * 32];   // [frag][lane], 32B per lane

    const int tid = threadIdx.x;

    // stage weights, swizzled into fragment layout: idx = frag*512 + lane*16 + j
    for (int idx = tid; idx < NFRAG * 512; idx += 256) {
        const int j    = idx & 15;
        const int ln   = (idx >> 4) & 31;
        const int f    = idx >> 9;
        const int kt   = f / NT;
        const int nt   = f - kt * NT;
        const int k    = kt * 32 + frag_k(j >> 1, j & 1, ln >> 4);
        const int n    = nt * 16 + (ln & 15);
        const float w  = (k < K0) ? B0[k * OUT + n] : B1[(k - K0) * OUT + n];
        ((__bf16*)sW)[idx] = f2bf(w);
    }
    __syncthreads();

    const int wave    = tid >> 5;
    const int lane    = tid & 31;
    const int half    = lane >> 4;
    const int r       = lane & 15;
    const int rowbase = blockIdx.x * 128 + wave * 16;
    int arow = rowbase + r;
    if (arow >= M) arow = M - 1;     // clamp for tail block (stores guarded)

    const v8f vzero = {0.f, 0.f, 0.f, 0.f, 0.f, 0.f, 0.f, 0.f};
    v8f acc[NT];
#pragma unroll
    for (int nt = 0; nt < NT; ++nt) acc[nt] = vzero;

#pragma unroll
    for (int kt = 0; kt < KT; ++kt) {
        const int kbase = kt * 32;
        v16bf a;
        if (kbase < K0) a = load_a_frag(A0 + (size_t)arow * K0, kbase, half);
        else            a = load_a_frag(A1 + (size_t)arow * K1, kbase - K0, half);

#pragma unroll
        for (int nt = 0; nt < NT; ++nt) {
            const v16bf b = sW[(kt * NT + nt) * 32 + lane];   // 2x ds_load_b128
            acc[nt] = __builtin_amdgcn_wmma_f32_16x16x32_bf16(
                false, a, false, b, (short)0, acc[nt], false, false);
        }
    }

    // Epilogue: bias + optional ReLU; C/D layout: VGPR i -> row i (+8 for half 1)
    const bool fulltile = (rowbase + 16 <= M);
#pragma unroll
    for (int nt = 0; nt < NT; ++nt) {
        const int col = nt * 16 + r;
        const float bv = bias[col];
#pragma unroll
        for (int i = 0; i < 8; ++i) {
            const int row = rowbase + i + half * 8;
            float v = acc[nt][i] + bv;
            if (RELU) v = v > 0.f ? v : 0.f;
            if (fulltile || row < M) {
                if (OBF) ((__bf16*)Cout)[(size_t)row * OUT + col] = f2bf(v);
                else     ((float*) Cout)[(size_t)row * OUT + col] = v;
            }
        }
    }
}

// ---------------------------------------------------------------------------
// Edge scatter:  agg[dst[e], :] += w[e] * h[src[e], :]   (H = 128, h in bf16)
// 16 threads per edge, 8 cols each: one 16B gather + 8 f32 hardware atomics.
// ---------------------------------------------------------------------------
__global__ __launch_bounds__(256)
void scatter_kernel(const __bf16* __restrict__ h, const int* __restrict__ src,
                    const int* __restrict__ dst, const float* __restrict__ w,
                    float* __restrict__ agg, int E)
{
    const long long t = (long long)blockIdx.x * 256 + threadIdx.x;
    const long long total = (long long)E * 16;
    if (t >= total) return;
    const int e = (int)(t >> 4);
    const int c = (int)(t & 15);
    const int s = src[e];
    const int d = dst[e];
    const float we = w[e];
    const v8bf hv = *(const v8bf*)(h + (size_t)s * 128 + c * 8);
    float* p = agg + (size_t)d * 128 + c * 8;
#pragma unroll
    for (int j = 0; j < 8; ++j)
        __hip_atomic_fetch_add(p + j, bf2f(hv[j]) * we,
                               __ATOMIC_RELAXED, __HIP_MEMORY_SCOPE_AGENT);
}

__global__ __launch_bounds__(256)
void zero_kernel(float* __restrict__ p, long long n)
{
    long long i = (long long)blockIdx.x * 256 + threadIdx.x;
    if (i < n) p[i] = 0.f;
}

// z = mu + eps * exp(0.5 * logvar)
__global__ __launch_bounds__(256)
void reparam_kernel(const float* __restrict__ mu, const float* __restrict__ lv,
                    const float* __restrict__ eps, float* __restrict__ z, long long n)
{
    long long i = (long long)blockIdx.x * 256 + threadIdx.x;
    if (i < n) z[i] = mu[i] + eps[i] * __expf(0.5f * lv[i]);
}

// ---------------------------------------------------------------------------
extern "C" void kernel_launch(void* const* d_in, const int* in_sizes, int n_in,
                              void* d_out, int out_size, void* d_ws, size_t ws_size,
                              hipStream_t stream)
{
    const float* x     = (const float*)d_in[0];
    const int*   eidx  = (const int*)d_in[1];
    const float* eattr = (const float*)d_in[2];
    const float* W1    = (const float*)d_in[3];
    const float* b1    = (const float*)d_in[4];
    const float* W2    = (const float*)d_in[5];
    const float* b2    = (const float*)d_in[6];
    const float* Wrel1 = (const float*)d_in[7];
    const float* brel1 = (const float*)d_in[8];
    const float* Wroot1= (const float*)d_in[9];
    const float* Wrel2 = (const float*)d_in[10];
    const float* brel2 = (const float*)d_in[11];
    const float* Wroot2= (const float*)d_in[12];
    const float* Wmu   = (const float*)d_in[13];
    const float* bmu   = (const float*)d_in[14];
    const float* Wlv   = (const float*)d_in[15];
    const float* blv   = (const float*)d_in[16];
    const float* Wd1   = (const float*)d_in[17];
    const float* bd1   = (const float*)d_in[18];
    const float* Wd2   = (const float*)d_in[19];
    const float* bd2   = (const float*)d_in[20];
    const float* eps   = (const float*)d_in[21];

    const int N = in_sizes[0] / 256;     // nodes
    const int E = in_sizes[2];           // edges
    const int* esrc = eidx;
    const int* edst = eidx + E;

    float* out = (float*)d_out;
    // x_recon region [0, N*256) f32 is written last -> reuse as bf16 scratch now
    __bf16* hA = (__bf16*)out;                  // N x 128 bf16
    __bf16* hB = hA + (size_t)N * 128;          // N x 128 bf16
    float*  hC = (float*)d_ws;                  // N x 128 f32 (agg)
    __bf16* hT = (__bf16*)d_ws;                 // N x 128 bf16 (decoder hidden)
    float* mu_out = out + (size_t)N * 256;      // N x 32 f32
    float* lv_out = mu_out + (size_t)N * 32;
    float* z_out  = lv_out + (size_t)N * 32;

    const int gRows = (N + 127) / 128;
    const int gZero = (int)(((long long)N * 128 + 255) / 256);
    const int gScat = (int)(((long long)E * 16 + 255) / 256);
    const int gRep  = (int)(((long long)N * 32 + 255) / 256);

    // 1) encoder MLP (x f32 -> bf16 activations)
    gemm_kernel<float,  float,  256, 0, 128, true,  true ><<<gRows, 256, 0, stream>>>(
        x, (const float*)nullptr, W1, nullptr, b1, hA, N);
    gemm_kernel<__bf16, __bf16, 128, 0, 128, true,  true ><<<gRows, 256, 0, stream>>>(
        hA, (const __bf16*)nullptr, W2, nullptr, b2, hB, N);

    // 2) GraphConv 1: agg = scatter(hB); hA = relu(agg@Wrel1 + hB@Wroot1 + brel1)
    zero_kernel<<<gZero, 256, 0, stream>>>(hC, (long long)N * 128);
    scatter_kernel<<<gScat, 256, 0, stream>>>(hB, esrc, edst, eattr, hC, E);
    gemm_kernel<float,  __bf16, 128, 128, 128, true, true ><<<gRows, 256, 0, stream>>>(
        hC, hB, Wrel1, Wroot1, brel1, hA, N);

    // 3) GraphConv 2
    zero_kernel<<<gZero, 256, 0, stream>>>(hC, (long long)N * 128);
    scatter_kernel<<<gScat, 256, 0, stream>>>(hA, esrc, edst, eattr, hC, E);
    gemm_kernel<float,  __bf16, 128, 128, 128, true, true ><<<gRows, 256, 0, stream>>>(
        hC, hA, Wrel2, Wroot2, brel2, hB, N);

    // 4) heads (f32 outputs) + reparameterize
    gemm_kernel<__bf16, __bf16, 128, 0, 32, false, false><<<gRows, 256, 0, stream>>>(
        hB, (const __bf16*)nullptr, Wmu, nullptr, bmu, mu_out, N);
    gemm_kernel<__bf16, __bf16, 128, 0, 32, false, false><<<gRows, 256, 0, stream>>>(
        hB, (const __bf16*)nullptr, Wlv, nullptr, blv, lv_out, N);
    reparam_kernel<<<gRep, 256, 0, stream>>>(mu_out, lv_out, eps, z_out, (long long)N * 32);

    // 5) decoder: hT = relu(z@Wd1+bd1) (bf16, in ws); x_recon = hT@Wd2+bd2 (f32)
    gemm_kernel<float,  float,   32, 0, 128, true,  true ><<<gRows, 256, 0, stream>>>(
        z_out, (const float*)nullptr, Wd1, nullptr, bd1, hT, N);
    gemm_kernel<__bf16, __bf16, 128, 0, 256, false, false><<<gRows, 256, 0, stream>>>(
        hT, (const __bf16*)nullptr, Wd2, nullptr, bd2, out, N);
}
